// FSLModel_35270271435100
// MI455X (gfx1250) — matrix operand order repaired and checked by reading, not compiled, not run on hardware
//
#include <hip/hip_runtime.h>
#include <hip/hip_bf16.h>
#include <math.h>

// ---------------- problem constants ----------------
#define N_NODES 1360   // multiple of 16
#define F_INN   1000
#define HIDD    2048
#define NHEAD   8
#define BB      16384
#define PF      1024   // F_IN padded to multiple of 32 (K dim)
#define PN      1376   // N padded to multiple of 32 (K dim)

typedef __bf16 bf16_t;
typedef bf16_t v16bf __attribute__((ext_vector_type(16)));
typedef bf16_t v8bf  __attribute__((ext_vector_type(8)));
typedef float  v8f   __attribute__((ext_vector_type(8)));

union Frag { v16bf v; v8bf h[2]; };

// ---------------- WMMA helpers (wave32, 16x16x32 bf16) ----------------
__device__ __forceinline__ v8f wmma_bf16(v16bf a, v16bf b, v8f c) {
  return __builtin_amdgcn_wmma_f32_16x16x32_bf16(false, a, false, b, (short)0, c,
                                                 false, false);
}

__device__ __forceinline__ v16bf load_frag2(const bf16_t* p, int gap) {
  Frag f;
  f.h[0] = *reinterpret_cast<const v8bf*>(p);
  f.h[1] = *reinterpret_cast<const v8bf*>(p + gap);
  return f.v;
}

// ---------------- generic bf16 GEMM: C[M][Nc] = A[M][K] * B, B fed as BT[Nc][K]
// REQUIRES Nc % 64 == 0 (no N edge -> branch-free K loop).
// MODE 0: C=acc (f32) plus transposed bf16 copy outT[Nc][ldT]
// MODE 1: C=acc
// MODE 2: C+=acc
// MODE 3: C=bias+acc (f32) plus row-major bf16 copy outB (same ldc)
template <int MODE>
__global__ __launch_bounds__(128) void k_gemm_bf16(
    const bf16_t* __restrict__ A, int lda, const bf16_t* __restrict__ BT, int ldb,
    int M, int K, float* __restrict__ C, int ldc,
    bf16_t* __restrict__ outT, int ldT, bf16_t* __restrict__ outB,
    const float* __restrict__ bias) {
  const int lane = threadIdx.x & 31;
  const int wave = threadIdx.x >> 5;
  const int m0 = blockIdx.x * 64 + wave * 16;
  if (m0 + 16 > M) return;  // uniform per wave; tiles are exact multiples of 16
  const int nBase = blockIdx.y * 64;
  const int kA = (lane >> 4) << 4;   // A half-K offset {0,16}
  const int kB = (lane >> 4) << 4;   // B half-K offset {0,16}
  const int bcol = lane & 15;

  const bf16_t* Ap = A + (size_t)(m0 + bcol) * lda + kA;
  const bf16_t* Bp0 = BT + (size_t)(nBase + 0 * 16 + bcol) * ldb + kB;
  const bf16_t* Bp1 = BT + (size_t)(nBase + 1 * 16 + bcol) * ldb + kB;
  const bf16_t* Bp2 = BT + (size_t)(nBase + 2 * 16 + bcol) * ldb + kB;
  const bf16_t* Bp3 = BT + (size_t)(nBase + 3 * 16 + bcol) * ldb + kB;

  v8f acc0 = {}, acc1 = {}, acc2 = {}, acc3 = {};
  for (int k0 = 0; k0 < K; k0 += 32) {
    v16bf a = load_frag2(Ap + k0, 8);
    acc0 = wmma_bf16(a, load_frag2(Bp0 + k0, 8), acc0);
    acc1 = wmma_bf16(a, load_frag2(Bp1 + k0, 8), acc1);
    acc2 = wmma_bf16(a, load_frag2(Bp2 + k0, 8), acc2);
    acc3 = wmma_bf16(a, load_frag2(Bp3 + k0, 8), acc3);
  }

  const int srow = m0 + ((lane >> 4) << 3);
  v8f accs[4] = {acc0, acc1, acc2, acc3};
#pragma unroll
  for (int j = 0; j < 4; ++j) {
    const int n = nBase + j * 16 + bcol;
    if (MODE == 0) {
      v8bf t;
#pragma unroll
      for (int r = 0; r < 8; ++r) {
        float v = accs[j][r];
        C[(size_t)(srow + r) * ldc + n] = v;
        t[r] = (bf16_t)v;
      }
      *reinterpret_cast<v8bf*>(outT + (size_t)n * ldT + srow) = t;  // contiguous
    } else if (MODE == 1) {
#pragma unroll
      for (int r = 0; r < 8; ++r) C[(size_t)(srow + r) * ldc + n] = accs[j][r];
    } else if (MODE == 2) {
#pragma unroll
      for (int r = 0; r < 8; ++r) {
        size_t i = (size_t)(srow + r) * ldc + n;
        C[i] += accs[j][r];
      }
    } else {
#pragma unroll
      for (int r = 0; r < 8; ++r) {
        size_t i = (size_t)(srow + r) * ldc + n;
        float v = bias[i] + accs[j][r];
        C[i] = v;
        outB[i] = (bf16_t)v;
      }
    }
  }
}

// ---------------- scores GEMM: C[M][Nc] = A[M][K] * BT[Nc][K]^T, Nc not a
// multiple of 64: loads use clamped row indices (branch-free K loop), stores
// are guarded per subtile.
__global__ __launch_bounds__(128) void k_gemm_nt(
    const bf16_t* __restrict__ A, const bf16_t* __restrict__ BT,
    float* __restrict__ C, int M, int Nc, int K) {
  const int lane = threadIdx.x & 31;
  const int wave = threadIdx.x >> 5;
  const int m0 = blockIdx.x * 64 + wave * 16;
  if (m0 + 16 > M) return;
  const int nBase = blockIdx.y * 64;
  const int kA = (lane >> 4) << 4;
  const int kB = (lane >> 4) << 4;
  const int bcol = lane & 15;

  bool nv[4];
  const bf16_t* Bp[4];
#pragma unroll
  for (int j = 0; j < 4; ++j) {
    int n0 = nBase + j * 16;
    nv[j] = (n0 + 16) <= Nc;
    int row = (nv[j] ? n0 : 0) + bcol;  // clamp loads to a valid tile
    Bp[j] = BT + (size_t)row * K + kB;
  }
  const bf16_t* Ap = A + (size_t)(m0 + bcol) * K + kA;

  v8f acc0 = {}, acc1 = {}, acc2 = {}, acc3 = {};
  for (int k0 = 0; k0 < K; k0 += 32) {
    __builtin_prefetch(Ap + k0 + 256, 0, 0);  // stream x ahead
    v16bf a = load_frag2(Ap + k0, 8);
    acc0 = wmma_bf16(a, load_frag2(Bp[0] + k0, 8), acc0);
    acc1 = wmma_bf16(a, load_frag2(Bp[1] + k0, 8), acc1);
    acc2 = wmma_bf16(a, load_frag2(Bp[2] + k0, 8), acc2);
    acc3 = wmma_bf16(a, load_frag2(Bp[3] + k0, 8), acc3);
  }

  const int srow = m0 + ((lane >> 4) << 3);
  v8f accs[4] = {acc0, acc1, acc2, acc3};
#pragma unroll
  for (int j = 0; j < 4; ++j) {
    if (!nv[j]) continue;  // edge tiles: discard
    const int n = nBase + j * 16 + bcol;
#pragma unroll
    for (int r = 0; r < 8; ++r) C[(size_t)(srow + r) * Nc + n] = accs[j][r];
  }
}

// scores_all = scores_v + 2*scores_t (linearity of x @ (v + 2t)^T)
__global__ void k_sall(const float* __restrict__ sv, const float* __restrict__ st,
                       float* __restrict__ sa, size_t n) {
  size_t i = (size_t)blockIdx.x * blockDim.x + threadIdx.x;
  size_t stride = (size_t)gridDim.x * blockDim.x;
  for (; i < n; i += stride) sa[i] = sv[i] + 2.0f * st[i];
}

// ---------------- block reductions ----------------
__device__ __forceinline__ float block_reduce_sum(float v, float* red) {
  int t = threadIdx.x;
  red[t] = v;
  __syncthreads();
  for (int s = 128; s > 0; s >>= 1) {
    if (t < s) red[t] += red[t + s];
    __syncthreads();
  }
  float r = red[0];
  __syncthreads();
  return r;
}
__device__ __forceinline__ float block_reduce_max(float v, float* red) {
  int t = threadIdx.x;
  red[t] = v;
  __syncthreads();
  for (int s = 128; s > 0; s >>= 1) {
    if (t < s) red[t] = fmaxf(red[t], red[t + s]);
    __syncthreads();
  }
  float r = red[0];
  __syncthreads();
  return r;
}

// row-normalize wv -> bf16, zero-padded K
__global__ void k_norm_wv(const float* __restrict__ wv, bf16_t* __restrict__ out) {
  __shared__ float red[256];
  const int n = blockIdx.x;
  const float* row = wv + (size_t)n * F_INN;
  float s = 0.f;
  for (int f = threadIdx.x; f < F_INN; f += 256) {
    float v = row[f];
    s += v * v;
  }
  s = block_reduce_sum(s, red);
  float inv = rsqrtf(s);
  for (int f = threadIdx.x; f < PF; f += 256)
    out[(size_t)n * PF + f] = (bf16_t)(f < F_INN ? row[f] * inv : 0.f);
}

// transpose+convert: src f32 (R x C) -> dst bf16 (C x PR), zero-padded rows >= R
__global__ void k_transpose_bf16(const float* __restrict__ src,
                                 bf16_t* __restrict__ dst, int R, int C, int PR) {
  __shared__ float tile[32][33];
  const int c0 = blockIdx.x * 32, r0 = blockIdx.y * 32;
  for (int i = threadIdx.y; i < 32; i += 8) {
    int r = r0 + i, c = c0 + threadIdx.x;
    tile[i][threadIdx.x] = (r < R && c < C) ? src[(size_t)r * C + c] : 0.f;
  }
  __syncthreads();
  for (int i = threadIdx.y; i < 32; i += 8) {
    int c = c0 + i, r = r0 + threadIdx.x;
    if (c < C && r < PR) dst[(size_t)c * PR + r] = (bf16_t)tile[threadIdx.x][i];
  }
}

__global__ void k_adj_bf(const float* __restrict__ adj, bf16_t* __restrict__ out) {
  size_t i = (size_t)blockIdx.x * 256 + threadIdx.x;
  const size_t tot = (size_t)N_NODES * PN;
  if (i >= tot) return;
  int n = (int)(i / PN), m = (int)(i % PN);
  float v = (m < N_NODES) ? adj[(size_t)n * N_NODES + m] : 0.f;
  out[i] = (bf16_t)v;
}

__global__ void k_cast_bf16(const float* __restrict__ s, bf16_t* __restrict__ d,
                            size_t n) {
  size_t i = (size_t)blockIdx.x * blockDim.x + threadIdx.x;
  size_t stride = (size_t)gridDim.x * blockDim.x;
  for (; i < n; i += stride) d[i] = (bf16_t)s[i];
}

// r1[n] = Wh[n]·a1, r2[n] = Wh[n]·a2
__global__ void k_r12(const float* __restrict__ Wh, const float* __restrict__ agh,
                      float* __restrict__ r1, float* __restrict__ r2) {
  __shared__ float red[256];
  const int n = blockIdx.x;
  const float* row = Wh + (size_t)n * HIDD;
  float s1 = 0.f, s2 = 0.f;
  for (int o = threadIdx.x; o < HIDD; o += 256) {
    float w = row[o];
    s1 += w * agh[o];
    s2 += w * agh[HIDD + o];
  }
  s1 = block_reduce_sum(s1, red);
  s2 = block_reduce_sum(s2, red);
  if (threadIdx.x == 0) {
    r1[n] = s1;
    r2[n] = s2;
  }
}

// masked leaky-relu softmax row; e[n][m] = r1[n]+r2[m]; writes bf16 att (adj*att)
__global__ void k_att(const float* __restrict__ r1, const float* __restrict__ r2,
                      const float* __restrict__ adj, bf16_t* __restrict__ att) {
  __shared__ float red[256];
  const int n = blockIdx.x;
  const float rn = r1[n];
  const float* arow = adj + (size_t)n * N_NODES;
  float mx = -3.0e38f;
  for (int m = threadIdx.x; m < N_NODES; m += 256) {
    if (arow[m] > 0.f) {
      float e = rn + r2[m];
      e = e > 0.f ? e : 0.2f * e;
      mx = fmaxf(mx, e);
    }
  }
  mx = block_reduce_max(mx, red);
  float s = 0.f;
  for (int m = threadIdx.x; m < N_NODES; m += 256) {
    if (arow[m] > 0.f) {
      float e = rn + r2[m];
      e = e > 0.f ? e : 0.2f * e;
      s += __expf(e - mx);
    }
  }
  s = block_reduce_sum(s, red);
  const float inv = 1.f / s;
  for (int m = threadIdx.x; m < PN; m += 256) {
    float v = 0.f;
    if (m < N_NODES && arow[m] > 0.f) {
      float e = rn + r2[m];
      e = e > 0.f ? e : 0.2f * e;
      v = __expf(e - mx) * inv;
    }
    att[(size_t)n * PN + m] = (bf16_t)v;
  }
}

// t_final = cw0*relu(t_origin) + (cw1/H)*t_trans_sum + cb ; also bf16 copy
__global__ void k_tfinal(const float* __restrict__ to, const float* __restrict__ tt,
                         const float* __restrict__ cw, const float* __restrict__ cb,
                         float* __restrict__ tf, bf16_t* __restrict__ tfb, size_t n) {
  size_t i = (size_t)blockIdx.x * 256 + threadIdx.x;
  if (i >= n) return;
  float v = cw[0] * fmaxf(to[i], 0.f) + (cw[1] * (1.0f / NHEAD)) * tt[i] + cb[0];
  tf[i] = v;
  tfb[i] = (bf16_t)v;
}

// ---------------- host launcher ----------------
extern "C" void kernel_launch(void* const* d_in, const int* in_sizes, int n_in,
                              void* d_out, int out_size, void* d_ws, size_t ws_size,
                              hipStream_t stream) {
  const float* x = (const float*)d_in[0];
  const float* wv = (const float*)d_in[1];
  const float* adj = (const float*)d_in[2];
  const float* Wg = (const float*)d_in[3];
  const float* ag = (const float*)d_in[4];
  const float* linW = (const float*)d_in[5];
  const float* cw = (const float*)d_in[6];
  const float* cb = (const float*)d_in[7];
  const float* W = (const float*)d_in[8];

  float* out = (float*)d_out;
  float* sv = out;
  float* st = sv + (size_t)BB * N_NODES;
  float* sa = st + (size_t)BB * N_NODES;
  float* tfo = sa + (size_t)BB * N_NODES;
  float* vfo = tfo + (size_t)N_NODES * HIDD;

  char* wsp = (char*)d_ws;
  size_t off = 0;
  auto take = [&](size_t bytes) -> void* {
    void* p = wsp + off;
    off = (off + bytes + 255) & ~(size_t)255;
    return p;
  };
  bf16_t* wv_bf = (bf16_t*)take((size_t)N_NODES * PF * 2);
  bf16_t* WgT = (bf16_t*)take((size_t)HIDD * PF * 2);
  bf16_t* linT = (bf16_t*)take((size_t)HIDD * PF * 2);
  bf16_t* Wt = (bf16_t*)take((size_t)HIDD * PN * 2);
  bf16_t* WhT = (bf16_t*)take((size_t)HIDD * PN * 2);
  float* Whf = (float*)take((size_t)N_NODES * HIDD * 4);
  bf16_t* attb = (bf16_t*)take((size_t)N_NODES * PN * 2);
  bf16_t* adjb = (bf16_t*)take((size_t)N_NODES * PN * 2);
  float* r12 = (float*)take((size_t)2 * N_NODES * 4);
  float* ttr = (float*)take((size_t)N_NODES * HIDD * 4);
  float* torig = (float*)take((size_t)N_NODES * HIDD * 4);
  bf16_t* tfb = (bf16_t*)take((size_t)N_NODES * HIDD * 2);
  bf16_t* vfb = (bf16_t*)take((size_t)N_NODES * HIDD * 2);
  bf16_t* xb = (bf16_t*)take((size_t)BB * HIDD * 2);

  hipMemsetAsync(ttr, 0, (size_t)N_NODES * HIDD * 4, stream);

  dim3 tb(32, 8);
  k_norm_wv<<<N_NODES, 256, 0, stream>>>(wv, wv_bf);
  k_transpose_bf16<<<dim3(HIDD / 32, PF / 32), tb, 0, stream>>>(linW, linT, F_INN,
                                                                HIDD, PF);
  k_transpose_bf16<<<dim3(HIDD / 32, PN / 32), tb, 0, stream>>>(W, Wt, N_NODES,
                                                                HIDD, PN);
  k_adj_bf<<<((size_t)N_NODES * PN + 255) / 256, 256, 0, stream>>>(adj, adjb);
  k_cast_bf16<<<4096, 256, 0, stream>>>(x, xb, (size_t)BB * HIDD);

  const dim3 gN((N_NODES + 63) / 64, HIDD / 64);  // (22, 32); Nc=2048 % 64 == 0
  // t_origin = wv_n @ lin_W
  k_gemm_bf16<1><<<gN, 128, 0, stream>>>(wv_bf, PF, linT, PF, N_NODES, PF, torig,
                                         HIDD, nullptr, 0, nullptr, nullptr);
  // v_final = W + adj @ W  (writes d_out + bf16 copy for scores GEMM)
  k_gemm_bf16<3><<<gN, 128, 0, stream>>>(adjb, PN, Wt, PN, N_NODES, PN, vfo, HIDD,
                                         nullptr, 0, vfb, W);

  for (int h = 0; h < NHEAD; ++h) {
    k_transpose_bf16<<<dim3(HIDD / 32, PF / 32), tb, 0, stream>>>(
        Wg + (size_t)h * F_INN * HIDD, WgT, F_INN, HIDD, PF);
    // Wh = wv_n @ Wg[h]  (f32 for attention logits + transposed bf16 for att GEMM)
    k_gemm_bf16<0><<<gN, 128, 0, stream>>>(wv_bf, PF, WgT, PF, N_NODES, PF, Whf,
                                           HIDD, WhT, PN, nullptr, nullptr);
    k_r12<<<N_NODES, 256, 0, stream>>>(Whf, ag + (size_t)h * 2 * HIDD, r12,
                                       r12 + N_NODES);
    k_att<<<N_NODES, 256, 0, stream>>>(r12, r12 + N_NODES, adj, attb);
    // t_trans += att @ Wh
    k_gemm_bf16<2><<<gN, 128, 0, stream>>>(attb, PN, WhT, PN, N_NODES, PN, ttr,
                                           HIDD, nullptr, 0, nullptr, nullptr);
  }

  k_tfinal<<<((size_t)N_NODES * HIDD + 255) / 256, 256, 0, stream>>>(
      torig, ttr, cw, cb, tfo, tfb, (size_t)N_NODES * HIDD);

  const dim3 gS(BB / 64, (N_NODES + 63) / 64);
  k_gemm_nt<<<gS, 128, 0, stream>>>(xb, vfb, sv, BB, N_NODES, HIDD);
  k_gemm_nt<<<gS, 128, 0, stream>>>(xb, tfb, st, BB, N_NODES, HIDD);
  k_sall<<<8192, 256, 0, stream>>>(sv, st, sa, (size_t)BB * N_NODES);
}
